// InvResMLP_11252814315650
// MI455X (gfx1250) — matrix operand order repaired and checked by reading, not compiled, not run on hardware
//
#include <hip/hip_runtime.h>
#include <cstddef>
#include <cstdint>

#define B_   8
#define N_   2048
#define C_   64
#define MID_ 256
#define KNN  16
#define CIN  67      /* 3 (dp) + 64 (f) input channels for GEMM1 */
#define KP   96      /* CIN padded up to a multiple of 32 */
#define PADH 8       /* LDS row padding (halves) to skew bank groups */
#define EPSV 1e-5f
#define RAD2 (0.15f * 0.15f)

typedef _Float16 half_t;
typedef __attribute__((ext_vector_type(16))) _Float16 v16h;
typedef __attribute__((ext_vector_type(8)))  float    v8f;

__device__ __forceinline__ v8f wmma_f16(const v16h& a, const v16h& b, const v8f& c) {
  // D = A(16x32 f16) * B(32x16 f16) + C(16x16 f32)
  return __builtin_amdgcn_wmma_f32_16x16x32_f16(false, a, false, b, (short)0, c,
                                                false, false);
}

// ---------------------------------------------------------------------------
// workspace zeroing (stats + mu/rs regions; d_ws is poisoned, not re-zeroed)
// ---------------------------------------------------------------------------
__global__ void zero_stats_kernel(float* ws, int n) {
  for (int t = threadIdx.x; t < n; t += blockDim.x) ws[t] = 0.0f;
}

// stats -> mu, rsqrt(var+eps)
__global__ void finalize_kernel(const float* __restrict__ stats,
                                float* __restrict__ mu, float* __restrict__ rs,
                                int ch, float inv_cnt) {
  int i = threadIdx.x;
  if (i < ch) {
    float m = stats[i] * inv_cnt;
    float v = stats[ch + i] * inv_cnt - m * m;
    mu[i] = m;
    rs[i] = rsqrtf(v + EPSV);
  }
}

// ---------------------------------------------------------------------------
// ball query: first KNN indices (ascending) with d2 < r^2; pad with first hit
// ---------------------------------------------------------------------------
__global__ void ball_query_kernel(const float* __restrict__ p, int* __restrict__ idx) {
  __shared__ float px[N_], py[N_], pz[N_];
  const int b = blockIdx.y;
  const int i = blockIdx.x * blockDim.x + threadIdx.x;
  for (int t = threadIdx.x; t < N_; t += blockDim.x) {
    const float* pp = p + ((size_t)b * N_ + t) * 3;
    px[t] = pp[0]; py[t] = pp[1]; pz[t] = pp[2];
  }
  __syncthreads();
  const float xi = px[i], yi = py[i], zi = pz[i];
  int nb[KNN];
  int cnt = 0;
  for (int j = 0; j < N_ && cnt < KNN; ++j) {
    float dx = px[j] - xi, dy = py[j] - yi, dz = pz[j] - zi;
    float d2 = dx * dx + dy * dy + dz * dz;
    if (d2 < RAD2) nb[cnt++] = j;
  }
  int f0 = (cnt > 0) ? nb[0] : (N_ - 1);   // self always qualifies; cnt>=1
  for (int t = cnt; t < KNN; ++t) nb[t] = f0;
  int* op = idx + ((size_t)b * N_ + i) * KNN;
  for (int t = 0; t < KNN; ++t) op[t] = nb[t];
}

// ---------------------------------------------------------------------------
// GEMM1: x[o,(n,k)] = w1[o,:] . [dp; f[:,idx]]   (64 x 67 x (B*N*K))
// one wave = one point n: its K=16 neighbors are the 16 WMMA columns.
// MODE 0: accumulate per-channel sum / sumsq of pre-BN x.
// MODE 1: BN + ReLU + max over K -> h (B,N,C) f16 (transposed for GEMM2).
// ---------------------------------------------------------------------------
template<int MODE>
__global__ __launch_bounds__(256)
void gemm1_kernel(const float* __restrict__ p, const float* __restrict__ f,
                  const float* __restrict__ w1, const int* __restrict__ idx,
                  float* __restrict__ stats,
                  const float* __restrict__ mu, const float* __restrict__ rs,
                  const float* __restrict__ gam, const float* __restrict__ bet,
                  half_t* __restrict__ h) {
  __shared__ half_t aw[C_][KP + PADH];          // w1 padded to K=96, f16
  __shared__ float  ssum[(MODE == 0) ? C_ : 1];
  __shared__ float  ssq [(MODE == 0) ? C_ : 1];
  const int tid = threadIdx.x, wave = tid >> 5, lane = tid & 31;

  for (int t = tid; t < C_ * KP; t += 256) {
    int o = t / KP, c = t % KP;
    aw[o][c] = (c < CIN) ? (half_t)w1[o * CIN + c] : (half_t)0.0f;
  }
  if (MODE == 0)
    for (int t = tid; t < C_; t += 256) { ssum[t] = 0.f; ssq[t] = 0.f; }
  __syncthreads();

  const int gp = blockIdx.x * 8 + wave;          // global point id
  const int b = gp / N_, n = gp % N_;
  const int k = lane & 15;                       // neighbor / column id
  const int j = idx[((size_t)b * N_ + n) * KNN + k];
  float dp0, dp1, dp2;
  {
    const float* pn = p + ((size_t)b * N_ + n) * 3;
    const float* pj = p + ((size_t)b * N_ + j) * 3;
    dp0 = pj[0] - pn[0]; dp1 = pj[1] - pn[1]; dp2 = pj[2] - pn[2];
  }
  const int base0 = (lane < 16) ? 0 : 16;  // A-fragment K base
  const int kbB   = (lane < 16) ? 0 : 16;  // B-fragment K base
  const int mrow  = lane & 15;
  const float* fb = f + (size_t)b * C_ * N_;

  v8f acc[4];
  for (int m = 0; m < 4; ++m)
    for (int e = 0; e < 8; ++e) acc[m][e] = 0.f;

  #pragma unroll
  for (int kk = 0; kk < 3; ++kk) {
    v16h bfrag;                                  // column = neighbor k
    #pragma unroll
    for (int i = 0; i < 16; ++i) {
      int c = kk * 32 + kbB + i;
      // branch-free: clamped unconditional gather + cndmask selects
      int cf = c - 3;
      cf = (cf < 0) ? 0 : ((cf > C_ - 1) ? C_ - 1 : cf);
      float fv = fb[(size_t)cf * N_ + j];
      float v = (c >= CIN) ? 0.0f
              : (c == 0) ? dp0 : (c == 1) ? dp1 : (c == 2) ? dp2 : fv;
      bfrag[i] = (half_t)v;
    }
    #pragma unroll
    for (int m = 0; m < 4; ++m) {
      v16h afrag;
      #pragma unroll
      for (int i = 0; i < 16; ++i) {
        int kc = kk * 32 + base0 + i;
        afrag[i] = aw[16 * m + mrow][kc];
      }
      acc[m] = wmma_f16(afrag, bfrag, acc[m]);
    }
  }

  if (MODE == 0) {
    #pragma unroll
    for (int m = 0; m < 4; ++m)
      #pragma unroll
      for (int r = 0; r < 8; ++r) {
        int o = 16 * m + ((lane < 16) ? r : 8 + r);
        float v = acc[m][r];
        atomicAdd(&ssum[o], v);
        atomicAdd(&ssq[o], v * v);
      }
    __syncthreads();
    for (int t = tid; t < C_; t += 256) {
      atomicAdd(&stats[t], ssum[t]);
      atomicAdd(&stats[C_ + t], ssq[t]);
    }
  } else {
    #pragma unroll
    for (int m = 0; m < 4; ++m)
      #pragma unroll
      for (int r = 0; r < 8; ++r) {
        int o = 16 * m + ((lane < 16) ? r : 8 + r);
        float v = (acc[m][r] - mu[o]) * rs[o] * gam[o] + bet[o];
        v = fmaxf(v, 0.0f);
        // max over the K=16 columns (lanes 0..15 / 16..31 of the half-wave)
        #pragma unroll
        for (int off = 1; off < 16; off <<= 1)
          v = fmaxf(v, __shfl_xor(v, off, 32));
        if ((lane & 15) == 0)
          h[((size_t)b * N_ + n) * C_ + o] = (half_t)v;   // transposed f16
      }
  }
}

// ---------------------------------------------------------------------------
// Generic channel GEMM over columns (b,n):  y[o,col] = w[o,:] . in[:,col]
// Input is f16 column-major (B, N, KIN): a lane's B-fragment is one
// contiguous 32-byte block -> single v16h (2x global_load_b128).
// MODE 0: stats(sum,sumsq).
// MODE 1: BN+ReLU -> f16 out, transposed (B, N, MOUT).
// MODE 2: BN + f residual + ReLU -> f32 out, standard (B, MOUT, N).
// ---------------------------------------------------------------------------
template<int MOUT, int KIN, int MODE>
__global__ __launch_bounds__(256)
void gemm_nk_kernel(const half_t* __restrict__ inp, const float* __restrict__ w,
                    float* __restrict__ stats,
                    const float* __restrict__ mu, const float* __restrict__ rs,
                    const float* __restrict__ gam, const float* __restrict__ bet,
                    void* __restrict__ outp, const float* __restrict__ fres) {
  constexpr int MT   = MOUT / 16;
  constexpr int TPW  = (MT > 8) ? 8 : MT;   // M-tiles per wave
  constexpr int NWG  = MT / TPW;            // waves per column group
  constexpr int GRPS = 8 / NWG;
  constexpr int COLS = GRPS * 16;
  __shared__ half_t aw[MOUT][KIN + PADH];
  __shared__ float  ssum[(MODE == 0) ? MOUT : 1];
  __shared__ float  ssq [(MODE == 0) ? MOUT : 1];
  const int tid = threadIdx.x, wave = tid >> 5, lane = tid & 31;

  for (int t = tid; t < MOUT * KIN; t += 256)
    aw[t / KIN][t % KIN] = (half_t)w[t];
  if (MODE == 0)
    for (int t = tid; t < MOUT; t += 256) { ssum[t] = 0.f; ssq[t] = 0.f; }
  __syncthreads();

  const int group = wave / NWG;
  const int msub  = wave % NWG;
  const int colg  = blockIdx.x * COLS + group * 16 + (lane & 15);
  const int bb = colg >> 11, nn = colg & (N_ - 1);
  const int base0 = (lane < 16) ? 0 : 16;
  const int kbB   = (lane < 16) ? 0 : 16;
  const int mrow  = lane & 15;
  const half_t* colp = inp + (size_t)colg * KIN;   // this lane's column

  v8f acc[TPW];
  for (int m = 0; m < TPW; ++m)
    for (int e = 0; e < 8; ++e) acc[m][e] = 0.f;

  #pragma unroll
  for (int kk = 0; kk < KIN / 32; ++kk) {
    // contiguous 16 halves = whole B-fragment for this lane
    v16h bfrag = *(const v16h*)(colp + kk * 32 + kbB);
    #pragma unroll
    for (int m = 0; m < TPW; ++m) {
      int mt = msub * TPW + m;
      v16h afrag;
      #pragma unroll
      for (int i = 0; i < 16; ++i) {
        int kc = kk * 32 + base0 + i;
        afrag[i] = aw[16 * mt + mrow][kc];
      }
      acc[m] = wmma_f16(afrag, bfrag, acc[m]);
    }
  }

  if (MODE == 0) {
    #pragma unroll
    for (int m = 0; m < TPW; ++m)
      #pragma unroll
      for (int r = 0; r < 8; ++r) {
        int o = 16 * (msub * TPW + m) + ((lane < 16) ? r : 8 + r);
        float v = acc[m][r];
        atomicAdd(&ssum[o], v);
        atomicAdd(&ssq[o], v * v);
      }
    __syncthreads();
    for (int t = tid; t < MOUT; t += 256) {
      atomicAdd(&stats[t], ssum[t]);
      atomicAdd(&stats[MOUT + t], ssq[t]);
    }
  } else {
    #pragma unroll
    for (int m = 0; m < TPW; ++m)
      #pragma unroll
      for (int r = 0; r < 8; ++r) {
        int o = 16 * (msub * TPW + m) + ((lane < 16) ? r : 8 + r);
        float v = (acc[m][r] - mu[o]) * rs[o] * gam[o] + bet[o];
        if (MODE == 1) {
          v = fmaxf(v, 0.0f);
          ((half_t*)outp)[(size_t)colg * MOUT + o] = (half_t)v;  // transposed
        } else {
          v += fres[((size_t)bb * MOUT + o) * N_ + nn];
          v = fmaxf(v, 0.0f);
          ((float*)outp)[((size_t)bb * MOUT + o) * N_ + nn] = v;
        }
      }
  }
}

// ---------------------------------------------------------------------------
extern "C" void kernel_launch(void* const* d_in, const int* in_sizes, int n_in,
                              void* d_out, int out_size, void* d_ws, size_t ws_size,
                              hipStream_t stream) {
  const float* p  = (const float*)d_in[0];
  const float* f  = (const float*)d_in[1];
  const float* w1 = (const float*)d_in[2];
  const float* g1 = (const float*)d_in[3];
  const float* b1 = (const float*)d_in[4];
  const float* w2 = (const float*)d_in[5];
  const float* g2 = (const float*)d_in[6];
  const float* b2 = (const float*)d_in[7];
  const float* w3 = (const float*)d_in[8];
  const float* g3 = (const float*)d_in[9];
  const float* b3 = (const float*)d_in[10];
  (void)in_sizes; (void)n_in; (void)out_size; (void)ws_size;

  float* wsf    = (float*)d_ws;
  float* stats1 = wsf;            // 128  (sum[64], sumsq[64])
  float* stats2 = wsf + 128;      // 512
  float* stats3 = wsf + 640;      // 128
  float* mu1 = wsf + 768,  *rs1 = wsf + 832;
  float* mu2 = wsf + 896,  *rs2 = wsf + 1152;
  float* mu3 = wsf + 1408, *rs3 = wsf + 1472;
  int*    idx = (int*)(wsf + 1536);                       // B*N*K ints (1 MB)
  half_t* h   = (half_t*)(wsf + 1536 + (size_t)B_ * N_ * KNN); // (B,N,C) f16 2MB
  half_t* y2  = h + (size_t)B_ * N_ * C_;                 // (B,N,MID) f16 8MB
  float*  out = (float*)d_out;

  zero_stats_kernel<<<dim3(1), dim3(256), 0, stream>>>(wsf, 1536);
  ball_query_kernel<<<dim3(N_ / 256, B_), dim3(256), 0, stream>>>(p, idx);

  // stage 1: gather + w1 GEMM
  gemm1_kernel<0><<<dim3(B_ * N_ / 8), dim3(256), 0, stream>>>(
      p, f, w1, idx, stats1, nullptr, nullptr, nullptr, nullptr, nullptr);
  finalize_kernel<<<dim3(1), dim3(256), 0, stream>>>(
      stats1, mu1, rs1, C_, 1.0f / (float)((size_t)B_ * N_ * KNN));
  gemm1_kernel<1><<<dim3(B_ * N_ / 8), dim3(256), 0, stream>>>(
      p, f, w1, idx, nullptr, mu1, rs1, g1, b1, h);

  // stage 2: w2 GEMM (256x64)
  gemm_nk_kernel<MID_, C_, 0><<<dim3((B_ * N_) / 64), dim3(256), 0, stream>>>(
      h, w2, stats2, nullptr, nullptr, nullptr, nullptr, nullptr, nullptr);
  finalize_kernel<<<dim3(1), dim3(256), 0, stream>>>(
      stats2, mu2, rs2, MID_, 1.0f / (float)((size_t)B_ * N_));
  gemm_nk_kernel<MID_, C_, 1><<<dim3((B_ * N_) / 64), dim3(256), 0, stream>>>(
      h, w2, nullptr, mu2, rs2, g2, b2, y2, nullptr);

  // stage 3: w3 GEMM (64x256) + residual
  gemm_nk_kernel<C_, MID_, 0><<<dim3((B_ * N_) / 128), dim3(256), 0, stream>>>(
      y2, w3, stats3, nullptr, nullptr, nullptr, nullptr, nullptr, nullptr);
  finalize_kernel<<<dim3(1), dim3(256), 0, stream>>>(
      stats3, mu3, rs3, C_, 1.0f / (float)((size_t)B_ * N_));
  gemm_nk_kernel<C_, MID_, 2><<<dim3((B_ * N_) / 128), dim3(256), 0, stream>>>(
      y2, w3, nullptr, mu3, rs3, g3, b3, out, f);
}
